// FlashAttention_67611375174105
// MI455X (gfx1250) — matrix operand order repaired and checked by reading, not compiled, not run on hardware
//
#include <hip/hip_runtime.h>
#include <hip/hip_bf16.h>

typedef __bf16 bf16;
typedef __attribute__((ext_vector_type(16))) __bf16 v16bf;
typedef __attribute__((ext_vector_type(8)))  __bf16 bf16x8;
typedef __attribute__((ext_vector_type(8)))  float  v8f;

// ---------------- helpers ----------------

__device__ inline bf16 f2bf(float f) {
  union { float f; unsigned u; } cv; cv.f = f;
  unsigned r = cv.u + 0x7FFFu + ((cv.u >> 16) & 1u);   // round-to-nearest-even
  unsigned short h = (unsigned short)(r >> 16);
  bf16 out;
  __builtin_memcpy(&out, &h, sizeof(out));
  return out;
}

// A fragment (16x32 MxK, bf16): lane L -> row m = L%16; half = L/16.
// elements 0..7  hold K = half*16 + e       (16B contiguous)
// elements 8..15 hold K = half*16 + e       (16B contiguous)
__device__ inline v16bf load_A(const bf16* A, int ld, int m0, int k0) {
  const int lane = threadIdx.x & 31;
  const int m = lane & 15, half = lane >> 4;
  const bf16* p = A + (size_t)(m0 + m) * ld + k0;
  bf16x8 lo = *(const bf16x8*)(p + half * 16);
  bf16x8 hi = *(const bf16x8*)(p + half * 16 + 8);
  v16bf a;
#pragma unroll
  for (int i = 0; i < 8; ++i) { a[i] = lo[i]; a[8 + i] = hi[i]; }
  return a;
}

// B fragment (32x16 KxN, bf16) for C = A * W^T:  B[k][n] = W[n][k].
// lane L -> col n = L%16; elements e hold K = (L/16)*16 + e (32B contiguous row of W)
__device__ inline v16bf load_B(const bf16* W, int ld, int n0, int k0) {
  const int lane = threadIdx.x & 31;
  const int n = lane & 15, half = lane >> 4;
  const bf16* p = W + (size_t)(n0 + n) * ld + k0 + half * 16;
  return *(const v16bf*)p;   // 32 bytes
}

__device__ inline v8f wmma_bf16(v16bf a, v16bf b, v8f c) {
  return __builtin_amdgcn_wmma_f32_16x16x32_bf16(
      /*neg_a=*/false, a, /*neg_b=*/false, b,
      /*c_mod=*/(short)0, c, /*reuse_a=*/false, /*reuse_b=*/false);
}

// ---------------- kernels ----------------

__global__ __launch_bounds__(256) void cvt_f32_bf16(const float* __restrict__ in,
                                                    bf16* __restrict__ out, int n) {
  int i = blockIdx.x * 256 + threadIdx.x;
  if (i < n) out[i] = f2bf(in[i]);
}

// qkv = x(bf16)[4096x1024] @ W_qkv^T(bf16)[3072x1024] + b_qkv
// scatter: Q,K -> [B,H,T,64] bf16 (Q pre-scaled by 1/sqrt(dk)), V -> transposed [B,H,64,T]
__global__ __launch_bounds__(256) void qkv_gemm_kernel(
    const bf16* __restrict__ X, const bf16* __restrict__ W,
    const float* __restrict__ bias,
    bf16* __restrict__ Qo, bf16* __restrict__ Ko, bf16* __restrict__ Vt) {
  const int lane = threadIdx.x & 31;
  const int wave = threadIdx.x >> 5;
  const int wm = wave & 3, wn = wave >> 2;
  const int m0 = blockIdx.y * 128 + wm * 32;
  const int n0 = blockIdx.x * 128 + wn * 64;

  v8f acc[2][4];
#pragma unroll
  for (int i = 0; i < 2; ++i)
#pragma unroll
    for (int t = 0; t < 4; ++t) acc[i][t] = (v8f)0.0f;

  for (int k0 = 0; k0 < 1024; k0 += 32) {
    v16bf a0 = load_A(X, 1024, m0, k0);
    v16bf a1 = load_A(X, 1024, m0 + 16, k0);
#pragma unroll
    for (int t = 0; t < 4; ++t) {
      v16bf b = load_B(W, 1024, n0 + 16 * t, k0);
      acc[0][t] = wmma_bf16(a0, b, acc[0][t]);
      acc[1][t] = wmma_bf16(a1, b, acc[1][t]);
    }
  }

  const int half = lane >> 4, nn = lane & 15;
#pragma unroll
  for (int t = 0; t < 4; ++t) {
    const int n = n0 + 16 * t + nn;
    const float bv = bias[n];
    const int which = n >> 10;          // 0:q 1:k 2:v  (uniform per tile)
    const int c = n & 1023, h = c >> 6, d = c & 63;
#pragma unroll
    for (int i = 0; i < 2; ++i) {
#pragma unroll
      for (int r = 0; r < 8; ++r) {
        const int m = m0 + 16 * i + r + 8 * half;
        const int bb = m >> 11, tt = m & 2047;
        const int zh = bb * 16 + h;
        const float v = acc[i][t][r] + bv;
        if (which == 0)      Qo[((size_t)zh * 2048 + tt) * 64 + d] = f2bf(v * 0.125f);
        else if (which == 1) Ko[((size_t)zh * 2048 + tt) * 64 + d] = f2bf(v);
        else                 Vt[((size_t)zh * 64 + d) * 2048 + tt] = f2bf(v);
      }
    }
  }
}

// Flash attention: grid.y = b*16+h (32), grid.x = T/64 (32). 4 waves x 16 query rows.
__global__ __launch_bounds__(128) void flash_kernel(
    const bf16* __restrict__ Q, const bf16* __restrict__ K,
    const bf16* __restrict__ Vt, bf16* __restrict__ att) {
  __shared__ __align__(32) bf16 Plds[4 * 16 * 64];
  const int lane = threadIdx.x & 31;
  const int wave = threadIdx.x >> 5;
  const int half = lane >> 4, nn = lane & 15;
  const int z = blockIdx.y, b = z >> 4, h = z & 15;
  const int t0 = blockIdx.x * 64 + wave * 16;

  const bf16* Qz = Q + (size_t)z * 2048 * 64;
  const bf16* Kz = K + (size_t)z * 2048 * 64;
  const bf16* Vz = Vt + (size_t)z * 64 * 2048;
  bf16* Pw = &Plds[wave * 16 * 64];

  // Q fragments are reused across the whole K/V loop (scale already folded in)
  v16bf aq0 = load_A(Qz, 64, t0, 0);
  v16bf aq1 = load_A(Qz, 64, t0, 32);

  v8f O[4];
#pragma unroll
  for (int dt = 0; dt < 4; ++dt) O[dt] = (v8f)0.0f;
  float mrow[8], lrow[8];
#pragma unroll
  for (int r = 0; r < 8; ++r) { mrow[r] = -3.0e38f; lrow[r] = 0.0f; }

  for (int s0 = 0; s0 < 2048; s0 += 64) {
    // S = Q @ K^T  (16 x 64 per wave)
    v8f S[4];
#pragma unroll
    for (int jt = 0; jt < 4; ++jt) S[jt] = (v8f)0.0f;
#pragma unroll
    for (int jt = 0; jt < 4; ++jt) {
      v16bf b0 = load_B(Kz, 64, s0 + 16 * jt, 0);
      v16bf b1 = load_B(Kz, 64, s0 + 16 * jt, 32);
      S[jt] = wmma_bf16(aq0, b0, S[jt]);
      S[jt] = wmma_bf16(aq1, b1, S[jt]);
    }
    // online softmax; each row m = r + 8*half lives in 16 lanes of one half
#pragma unroll
    for (int r = 0; r < 8; ++r) {
      float v = fmaxf(fmaxf(S[0][r], S[1][r]), fmaxf(S[2][r], S[3][r]));
      v = fmaxf(v, __shfl_xor(v, 1, 32));
      v = fmaxf(v, __shfl_xor(v, 2, 32));
      v = fmaxf(v, __shfl_xor(v, 4, 32));
      v = fmaxf(v, __shfl_xor(v, 8, 32));
      const float mnew = fmaxf(mrow[r], v);
      const float alpha = __expf(mrow[r] - mnew);
      mrow[r] = mnew;
      float sum = 0.0f;
#pragma unroll
      for (int jt = 0; jt < 4; ++jt) {
        const float p = __expf(S[jt][r] - mnew);
        S[jt][r] = p;
        sum += p;
      }
      sum += __shfl_xor(sum, 1, 32);
      sum += __shfl_xor(sum, 2, 32);
      sum += __shfl_xor(sum, 4, 32);
      sum += __shfl_xor(sum, 8, 32);
      lrow[r] = lrow[r] * alpha + sum;
#pragma unroll
      for (int dt = 0; dt < 4; ++dt) O[dt][r] *= alpha;
    }
    // stage P (bf16) through LDS: C-layout store -> A-layout reload
#pragma unroll
    for (int jt = 0; jt < 4; ++jt)
#pragma unroll
      for (int r = 0; r < 8; ++r)
        Pw[(r + 8 * half) * 64 + jt * 16 + nn] = f2bf(S[jt][r]);
    asm volatile("s_wait_dscnt 0x0" ::: "memory");
    v16bf ap0 = load_A(Pw, 64, 0, 0);
    v16bf ap1 = load_A(Pw, 64, 0, 32);
    // O += P @ V  (V is transposed: B[k=s][n=d] = Vt[d][s], contiguous per lane)
#pragma unroll
    for (int dt = 0; dt < 4; ++dt) {
      v16bf bv0 = load_B(Vz, 2048, 16 * dt, s0);
      v16bf bv1 = load_B(Vz, 2048, 16 * dt, s0 + 32);
      O[dt] = wmma_bf16(ap0, bv0, O[dt]);
      O[dt] = wmma_bf16(ap1, bv1, O[dt]);
    }
  }
  // normalize and write att as [B,T,H*dk] bf16
#pragma unroll
  for (int r = 0; r < 8; ++r) {
    const float inv = 1.0f / lrow[r];
    const int t = t0 + r + 8 * half;
#pragma unroll
    for (int dt = 0; dt < 4; ++dt) {
      const int d = dt * 16 + nn;
      att[((size_t)(b * 2048 + t)) * 1024 + h * 64 + d] = f2bf(O[dt][r] * inv);
    }
  }
}

// out = att(bf16)[4096x1024] @ W_out^T(bf16)[1024x1024] + b_out  -> f32
__global__ __launch_bounds__(256) void out_gemm_kernel(
    const bf16* __restrict__ X, const bf16* __restrict__ W,
    const float* __restrict__ bias, float* __restrict__ out) {
  const int lane = threadIdx.x & 31;
  const int wave = threadIdx.x >> 5;
  const int wm = wave & 3, wn = wave >> 2;
  const int m0 = blockIdx.y * 128 + wm * 32;
  const int n0 = blockIdx.x * 128 + wn * 64;

  v8f acc[2][4];
#pragma unroll
  for (int i = 0; i < 2; ++i)
#pragma unroll
    for (int t = 0; t < 4; ++t) acc[i][t] = (v8f)0.0f;

  for (int k0 = 0; k0 < 1024; k0 += 32) {
    v16bf a0 = load_A(X, 1024, m0, k0);
    v16bf a1 = load_A(X, 1024, m0 + 16, k0);
#pragma unroll
    for (int t = 0; t < 4; ++t) {
      v16bf b = load_B(W, 1024, n0 + 16 * t, k0);
      acc[0][t] = wmma_bf16(a0, b, acc[0][t]);
      acc[1][t] = wmma_bf16(a1, b, acc[1][t]);
    }
  }

  const int half = lane >> 4, nn = lane & 15;
#pragma unroll
  for (int t = 0; t < 4; ++t) {
    const int n = n0 + 16 * t + nn;
    const float bv = bias[n];
#pragma unroll
    for (int i = 0; i < 2; ++i) {
#pragma unroll
      for (int r = 0; r < 8; ++r) {
        const int m = m0 + 16 * i + r + 8 * half;
        out[(size_t)m * 1024 + n] = acc[i][t][r] + bv;
      }
    }
  }
}

// ---------------- launch ----------------

extern "C" void kernel_launch(void* const* d_in, const int* in_sizes, int n_in,
                              void* d_out, int out_size, void* d_ws, size_t ws_size,
                              hipStream_t stream) {
  (void)in_sizes; (void)n_in; (void)out_size; (void)ws_size;
  const float* x     = (const float*)d_in[0];  // [2,2048,1024]
  const float* W_qkv = (const float*)d_in[1];  // [3072,1024]
  const float* b_qkv = (const float*)d_in[2];  // [3072]
  const float* W_out = (const float*)d_in[3];  // [1024,1024]
  const float* b_out = (const float*)d_in[4];  // [1024]
  float* out = (float*)d_out;                  // [2,2048,1024]

  char* ws = (char*)d_ws;
  const size_t MB = 1024 * 1024;
  bf16* xb    = (bf16*)(ws +  0 * MB);  // 4,194,304 bf16 (8 MB)
  bf16* wqkvb = (bf16*)(ws +  8 * MB);  // 3,145,728 bf16 (6 MB)
  bf16* woutb = (bf16*)(ws + 14 * MB);  // 1,048,576 bf16 (2 MB)
  bf16* q     = (bf16*)(ws + 16 * MB);  // [2,16,2048,64] (8 MB)
  bf16* k     = (bf16*)(ws + 24 * MB);  // [2,16,2048,64] (8 MB)
  bf16* vt    = (bf16*)(ws + 32 * MB);  // [2,16,64,2048] (8 MB)
  bf16* att   = (bf16*)(ws + 40 * MB);  // [2,2048,1024]  (8 MB)  total 48 MB

  cvt_f32_bf16<<<(4194304 + 255) / 256, 256, 0, stream>>>(x, xb, 4194304);
  cvt_f32_bf16<<<(3145728 + 255) / 256, 256, 0, stream>>>(W_qkv, wqkvb, 3145728);
  cvt_f32_bf16<<<(1048576 + 255) / 256, 256, 0, stream>>>(W_out, woutb, 1048576);

  qkv_gemm_kernel<<<dim3(24, 32), 256, 0, stream>>>(xb, wqkvb, b_qkv, q, k, vt);
  flash_kernel<<<dim3(32, 32), 128, 0, stream>>>(q, k, vt, att);
  out_gemm_kernel<<<dim3(8, 32), 256, 0, stream>>>(att, woutb, b_out, out);
}
